// RGM_21431886807532
// MI455X (gfx1250) — matrix operand order repaired and checked by reading, not compiled, not run on hardware
//
#include <hip/hip_runtime.h>
#include <hip/hip_bf16.h>

// ---------------------------------------------------------------------------
// RNN forward for MI455X (gfx1250, wave32, WMMA, TDM).
//   B=16, T=4096, I=256, L=512
// Phase 1: xproj = x @ Wx^T + b   (65536 x 256 x 512 GEMM, memory-bound)
//          -> written directly into d_out[b][t][:]
// Phase 2: 8 persistent workgroups scan t=0..T-1:
//          z = tanh(xproj_t + z @ Wz^T), in-place update of d_out.
//          bf16 state ping-pongs through d_ws; each step the Tensor Data
//          Mover (tensor_load_to_lds) pulls the 16KB state into LDS once per
//          WGP; device-scope atomic barrier + agent fences between steps.
// ---------------------------------------------------------------------------

typedef __attribute__((ext_vector_type(16))) __bf16   v16bf;
typedef __attribute__((ext_vector_type(8)))  float    v8f;
typedef __attribute__((ext_vector_type(4)))  unsigned u32x4;
typedef __attribute__((ext_vector_type(8)))  unsigned u32x8;

#define RNN_B 16
#define RNN_T 4096
#define RNN_I 256
#define RNN_L 512
#define NWG_SCAN 8   // persistent workgroups in phase 2 (512 cols / 64)

union FragAB {            // one 16x32 (or 32x16) bf16 WMMA operand = 32 bytes/lane
    v16bf v;
    uint4 q[2];
};

// ---------------------------------------------------------------------------
// Phase 1: xproj GEMM.  Block = 256 threads (8 waves), wave: 16 rows x 64 cols.
// Wx tile staged in LDS as bf16, k-major: B fragment = contiguous 32B LDS read.
// ---------------------------------------------------------------------------
__global__ void rgm_xproj_gemm(const float* __restrict__ x,
                               const float* __restrict__ Wx,
                               const float* __restrict__ bias,
                               float* __restrict__ out)
{
    __shared__ __align__(32) __bf16 ldsB[RNN_I * 64];   // 32 KB: ldsB[n*RNN_I + k] = Wx[n0+n][k]

    const int tid  = threadIdx.x;
    const int wave = tid >> 5;
    const int lane = tid & 31;
    const int row  = lane & 15;
    const int hi   = lane >> 4;            // 0: K 0..15 ; 1: K 16..31

    const int n0 = blockIdx.y * 64;
    const int m0 = blockIdx.x * 128 + wave * 16;

    for (int i = tid; i < RNN_I * 64; i += 256) {
        int k = i & (RNN_I - 1);
        int n = i >> 8;
        ldsB[n * RNN_I + k] = (__bf16)Wx[(size_t)(n0 + n) * RNN_I + k];
    }
    __syncthreads();

    v8f acc[4] = {v8f{}, v8f{}, v8f{}, v8f{}};
    const float* xr = x + (size_t)(m0 + row) * RNN_I + hi * 16;

    for (int k0 = 0; k0 < RNN_I; k0 += 32) {
        float4 a0 = *(const float4*)(xr + k0);
        float4 a1 = *(const float4*)(xr + k0 + 4);
        float4 a2 = *(const float4*)(xr + k0 + 8);
        float4 a3 = *(const float4*)(xr + k0 + 12);
        v16bf a;
        a[0]=(__bf16)a0.x;  a[1]=(__bf16)a0.y;  a[2]=(__bf16)a0.z;  a[3]=(__bf16)a0.w;
        a[4]=(__bf16)a1.x;  a[5]=(__bf16)a1.y;  a[6]=(__bf16)a1.z;  a[7]=(__bf16)a1.w;
        a[8]=(__bf16)a2.x;  a[9]=(__bf16)a2.y;  a[10]=(__bf16)a2.z; a[11]=(__bf16)a2.w;
        a[12]=(__bf16)a3.x; a[13]=(__bf16)a3.y; a[14]=(__bf16)a3.z; a[15]=(__bf16)a3.w;

        #pragma unroll
        for (int nt = 0; nt < 4; ++nt) {
            v16bf b = *(const v16bf*)&ldsB[(nt * 16 + row) * RNN_I + k0 + hi * 16];
            acc[nt] = __builtin_amdgcn_wmma_f32_16x16x32_bf16(
                false, a, false, b, (short)0, acc[nt], false, false);
        }
    }

    #pragma unroll
    for (int nt = 0; nt < 4; ++nt) {
        const int   n  = n0 + nt * 16 + (lane & 15);
        const float bv = bias[n];
        #pragma unroll
        for (int r = 0; r < 8; ++r) {
            int m = m0 + r + hi * 8;
            out[(size_t)m * RNN_L + n] = acc[nt][r] + bv;
        }
    }
}

// ---------------------------------------------------------------------------
// Init: bf16 state buffer 0 <- z0, buffer 1 <- 0, barrier counter <- 0.
// ---------------------------------------------------------------------------
__global__ void rgm_scan_init(const float* __restrict__ z0,
                              __bf16* __restrict__ zb0,
                              __bf16* __restrict__ zb1,
                              unsigned* __restrict__ bar)
{
    int i = blockIdx.x * blockDim.x + threadIdx.x;
    if (i < RNN_B * RNN_L) {
        zb0[i] = (__bf16)z0[i];
        zb1[i] = (__bf16)0.0f;
    }
    if (i == 0) *bar = 0u;
}

// ---------------------------------------------------------------------------
// TDM descriptor group 0 for a 16KB (8192 x bf16) 1-D state tile -> LDS.
// Layout per CDNA5 ISA:
//   w0: count=1 (bits 1:0), no gather/restore
//   w1: lds_addr (bytes)
//   w2: global_addr[31:0]
//   w3: global_addr[56:32] | type=2 (bits 31:30)
// ---------------------------------------------------------------------------
__device__ inline u32x4 tdm_group0(const __bf16* gptr, unsigned lds_off)
{
    unsigned long long ga = (unsigned long long)(uintptr_t)gptr;
    u32x4 g;
    g[0] = 1u;
    g[1] = lds_off;
    g[2] = (unsigned)ga;
    g[3] = (unsigned)((ga >> 32) & 0x1FFFFFFull) | (2u << 30);
    return g;
}

// ---------------------------------------------------------------------------
// Phase 2: persistent sequential scan. 8 blocks x 128 threads (4 waves).
// Block wg owns cols [wg*64, wg*64+64), wave -> one 16x16 tile, K=512 ->
// 16 bf16 WMMAs/step split over two accumulator chains.
// LDS: 64 KB Wz slice + 16 KB state tile (legal: CDNA5 has 320 KB/WGP).
// ---------------------------------------------------------------------------
__global__ void rgm_scan(const float* __restrict__ Wz,
                         float* __restrict__ out,
                         __bf16* __restrict__ zb0,
                         __bf16* __restrict__ zb1,
                         unsigned* __restrict__ bar)
{
    __shared__ __align__(32) __bf16 ldsB[RNN_L * 64];      // 64 KB: ldsB[n*RNN_L+k] = Wz[wg*64+n][k]
    __shared__ __align__(16) __bf16 ldsZ[RNN_B * RNN_L];   // 16 KB: state tile (TDM target)

    const int wg   = blockIdx.x;
    const int tid  = threadIdx.x;
    const int wave = tid >> 5;
    const int lane = tid & 31;
    const int row  = lane & 15;
    const int hi   = lane >> 4;
    const int n0   = wg * 64 + wave * 16;

    // --- stage Wz slice (transposed, bf16) into LDS, once ---
    for (int i = tid; i < RNN_L * 64; i += 128) {
        int k = i & (RNN_L - 1);
        int n = i >> 9;
        ldsB[n * RNN_L + k] = (__bf16)Wz[(size_t)(wg * 64 + n) * RNN_L + k];
    }
    __syncthreads();

    // --- TDM descriptors (uniform, live in SGPRs) ---
    // Flat LDS pointers carry the byte offset in the low 32 bits.
    const unsigned ldsZ_off = (unsigned)(uintptr_t)&ldsZ[0];
    const u32x4 g0_even = tdm_group0(zb0, ldsZ_off);   // t even reads zb0
    const u32x4 g0_odd  = tdm_group0(zb1, ldsZ_off);   // t odd  reads zb1
    u32x8 g1;
    g1[0] = 1u << 16;                      // data_size=1 (2 bytes); no mask/iter/pad
    g1[1] = (RNN_B * RNN_L) << 16;         // tensor_dim0 lo16 = 8192
    g1[2] = 1u << 16;                      // tensor_dim0 hi=0 | tensor_dim1=1
    g1[3] = (RNN_B * RNN_L) << 16;         // tensor_dim1 hi=0 | tile_dim0 = 8192
    g1[4] = 0u;                            // tile_dim1/2 unused
    g1[5] = (RNN_B * RNN_L);               // tensor_dim0_stride lo32
    g1[6] = 0u;
    g1[7] = 0u;

    for (int t = 0; t < RNN_T; ++t) {
        __bf16* zout = (t & 1) ? zb0 : zb1;

        // --- Tensor Data Mover: DMA 16KB state tile into LDS (one per WGP) ---
        if (tid < 32) {                       // wave-uniform: wave 0 only
            const u32x4 g0 = (t & 1) ? g0_odd : g0_even;
            asm volatile("tensor_load_to_lds %0, %1"
                         :: "s"(g0), "s"(g1) : "memory");
            __builtin_amdgcn_s_wait_tensorcnt(0);
        }
        __syncthreads();                      // publish ldsZ to all 4 waves

        // --- z @ Wz^T tile: two independent WMMA accumulator chains ---
        v8f acc0 = v8f{};
        v8f acc1 = v8f{};
        const __bf16* zr = ldsZ + (size_t)row * RNN_L + hi * 16;

        #pragma unroll
        for (int kc = 0; kc < 16; ++kc) {
            const int k0 = kc * 32;
            FragAB a;
            a.q[0] = *(const uint4*)(zr + k0);        // K = k0+base .. +7
            a.q[1] = *(const uint4*)(zr + k0 + 8);    // K = k0+base+8 .. +15
            v16bf b = *(const v16bf*)&ldsB[(wave * 16 + row) * RNN_L + k0 + hi * 16];
            if (kc & 1)
                acc1 = __builtin_amdgcn_wmma_f32_16x16x32_bf16(
                    false, a.v, false, b, (short)0, acc1, false, false);
            else
                acc0 = __builtin_amdgcn_wmma_f32_16x16x32_bf16(
                    false, a.v, false, b, (short)0, acc0, false, false);
        }

        // --- z_new = tanh(xproj + acc); update out in place; bf16 copy ---
        const int n = n0 + (lane & 15);
        #pragma unroll
        for (int r = 0; r < 8; ++r) {
            int    m    = r + hi * 8;                       // batch row
            size_t oidx = ((size_t)m * RNN_T + t) * RNN_L + n;
            float  v    = tanhf(out[oidx] + (acc0[r] + acc1[r]));
            out[oidx]   = v;
            zout[m * RNN_L + n] = (__bf16)v;
            if (t == RNN_T - 1)                             // final hidden state
                out[(size_t)RNN_B * RNN_T * RNN_L + m * RNN_L + n] = v;
        }

        // ---- device-scope split barrier across the 8 persistent workgroups ----
        __builtin_amdgcn_fence(__ATOMIC_RELEASE, "agent");  // flush this wave's stores
        __syncthreads();
        if (tid == 0) {
            __hip_atomic_fetch_add(bar, 1u, __ATOMIC_RELAXED, __HIP_MEMORY_SCOPE_AGENT);
            const unsigned target = (unsigned)NWG_SCAN * (unsigned)(t + 1);
            while (__hip_atomic_load(bar, __ATOMIC_RELAXED, __HIP_MEMORY_SCOPE_AGENT) < target)
                __builtin_amdgcn_s_sleep(1);
        }
        __syncthreads();
        __builtin_amdgcn_fence(__ATOMIC_ACQUIRE, "agent");  // invalidate stale WGP$ lines
    }
}

// ---------------------------------------------------------------------------
extern "C" void kernel_launch(void* const* d_in, const int* in_sizes, int n_in,
                              void* d_out, int out_size, void* d_ws, size_t ws_size,
                              hipStream_t stream)
{
    (void)in_sizes; (void)n_in; (void)out_size; (void)ws_size;

    const float* x    = (const float*)d_in[0];   // (B,T,I)
    const float* z0   = (const float*)d_in[1];   // (B,L)
    const float* Wx   = (const float*)d_in[2];   // (L,I)
    const float* Wz   = (const float*)d_in[3];   // (L,L)
    const float* bias = (const float*)d_in[4];   // (L,)
    float*       out  = (float*)d_out;           // (B,T,L) ++ (B,L)

    __bf16*   zb0 = (__bf16*)d_ws;                          // 16 KB
    __bf16*   zb1 = zb0 + RNN_B * RNN_L;                    // 16 KB
    unsigned* bar = (unsigned*)((char*)d_ws + 32 * 1024);   // 4 B

    rgm_xproj_gemm<<<dim3(512, 8), 256, 0, stream>>>(x, Wx, bias, out);
    rgm_scan_init<<<dim3((RNN_B * RNN_L + 255) / 256), 256, 0, stream>>>(z0, zb0, zb1, bar);
    rgm_scan<<<dim3(NWG_SCAN), 128, 0, stream>>>(Wz, out, zb0, zb1, bar);
}
